// MeanShift_88424786690284
// MI455X (gfx1250) — matrix operand order repaired and checked by reading, not compiled, not run on hardware
//
#include <hip/hip_runtime.h>

// ---------------------------------------------------------------------------
// MeanShift kNN loss for MI455X (gfx1250, wave32, WMMA bf16 16x16x32)
// B=512 queries, D=512 dims, K=128000 bank rows, top-k = 5.
// ---------------------------------------------------------------------------

#define BQ 512       // batch (queries)
#define DQ 512       // feature dim
#define KQ 128000    // bank rows
#define TOPK 5
#define NSLICE 100            // K-dimension slices
#define SLICE_ROWS (KQ / NSLICE)      // 1280 bank rows per slice
#define TILES_PER_SLICE (SLICE_ROWS / 16)  // 80
#define WAVES 4
#define TILES_PER_WAVE (TILES_PER_SLICE / WAVES)  // 20

typedef __attribute__((ext_vector_type(16))) __bf16          v16bf;
typedef __attribute__((ext_vector_type(16))) unsigned short  v16us;
typedef __attribute__((ext_vector_type(8)))  unsigned short  v8us;
typedef __attribute__((ext_vector_type(8)))  float           v8f;

__device__ __forceinline__ unsigned short f2bf(float f) {
  unsigned u = __builtin_bit_cast(unsigned, f);
  u += 0x7fffu + ((u >> 16) & 1u);       // round-to-nearest-even
  return (unsigned short)(u >> 16);
}
__device__ __forceinline__ v16bf as_bf(v16us v) {
  return __builtin_bit_cast(v16bf, v);
}

// Register-resident top-5 (named fields -> no scratch spills).
struct Top5 {
  float t0, t1, t2, t3, t4;   // target distances (selection key, keep 5 smallest)
  float q0, q1, q2, q3, q4;   // query distances
  int   c0, c1, c2, c3, c4;   // bank row indices
};
__device__ __forceinline__ void top5_init(Top5& s) {
  s.t0 = s.t1 = s.t2 = s.t3 = s.t4 = 3.0e38f;
  s.q0 = s.q1 = s.q2 = s.q3 = s.q4 = 0.f;
  s.c0 = s.c1 = s.c2 = s.c3 = s.c4 = 0;
}
__device__ __forceinline__ void top5_ins(Top5& s, float tv, float qv, int cv) {
  float m = s.t0; int w = 0;
  if (s.t1 > m) { m = s.t1; w = 1; }
  if (s.t2 > m) { m = s.t2; w = 2; }
  if (s.t3 > m) { m = s.t3; w = 3; }
  if (s.t4 > m) { m = s.t4; w = 4; }
  if (tv < m) {
    if      (w == 0) { s.t0 = tv; s.q0 = qv; s.c0 = cv; }
    else if (w == 1) { s.t1 = tv; s.q1 = qv; s.c1 = cv; }
    else if (w == 2) { s.t2 = tv; s.q2 = qv; s.c2 = cv; }
    else if (w == 3) { s.t3 = tv; s.q3 = qv; s.c3 = cv; }
    else             { s.t4 = tv; s.q4 = qv; s.c4 = cv; }
  }
}

// ---------------------------------------------------------------------------
// Kernel 1: inverse L2 norms for bank rows [0,K), tgt rows [K,K+B), q rows
// [K+B,K+2B). One wave per row (wave32).
// ---------------------------------------------------------------------------
__global__ __launch_bounds__(256)
void inv_norms_kernel(const float* __restrict__ bank,
                      const float* __restrict__ tgt,
                      const float* __restrict__ qf,
                      float* __restrict__ inv) {
  const int row  = blockIdx.x * 8 + (threadIdx.x >> 5);
  const int lane = threadIdx.x & 31;
  if (row >= KQ + 2 * BQ) return;
  const float* src;
  if      (row < KQ)       src = bank + (size_t)row * DQ;
  else if (row < KQ + BQ)  src = tgt  + (size_t)(row - KQ) * DQ;
  else                     src = qf   + (size_t)(row - KQ - BQ) * DQ;
  const float4* s4 = (const float4*)src;
  float s = 0.f;
#pragma unroll
  for (int i = 0; i < 4; ++i) {
    float4 v = s4[lane + 32 * i];
    s += v.x * v.x + v.y * v.y + v.z * v.z + v.w * v.w;
  }
#pragma unroll
  for (int o = 16; o > 0; o >>= 1) s += __shfl_xor(s, o, 32);
  if (lane == 0) inv[row] = rsqrtf(s + 1e-12f);
}

// ---------------------------------------------------------------------------
// Kernel 2: bank f32 -> normalized bf16 (131 MB, fits in 192 MB L2).
// ---------------------------------------------------------------------------
__global__ __launch_bounds__(256)
void cvt_bank_kernel(const float* __restrict__ bank,
                     const float* __restrict__ inv,
                     unsigned short* __restrict__ out) {
  const size_t idx = ((size_t)blockIdx.x * 256 + threadIdx.x) * 4;
  if (idx >= (size_t)KQ * DQ) return;
  const int row = (int)(idx >> 9);
  const float s = inv[row];
  float4 v = *(const float4*)(bank + idx);
  uint2 p;
  p.x = (unsigned)f2bf(v.x * s) | ((unsigned)f2bf(v.y * s) << 16);
  p.y = (unsigned)f2bf(v.z * s) | ((unsigned)f2bf(v.w * s) << 16);
  *(uint2*)(out + idx) = p;
}

// ---------------------------------------------------------------------------
// Kernel 3: fused dual-GEMM + streaming top-5.
// grid = (32 row-tiles, 100 slices); block = 128 threads = 4 waves.
// ---------------------------------------------------------------------------
__global__ __launch_bounds__(128)
void meanshift_main_kernel(const float* __restrict__ tgt,
                           const float* __restrict__ qf,
                           const unsigned short* __restrict__ bankbf,
                           const float* __restrict__ inv,
                           float* __restrict__ candT,
                           float* __restrict__ candQ,
                           int*   __restrict__ candC) {
  __shared__ unsigned short sAt[16 * DQ];   // normalized bf16 tgt rows (16 KB)
  __shared__ unsigned short sAq[16 * DQ];   // normalized bf16 q rows   (16 KB)
  __shared__ float sDist[WAVES][512];       // per-wave 16x16 t + q tiles (8 KB)
  __shared__ float mT[WAVES][16][TOPK];
  __shared__ float mQ[WAVES][16][TOPK];
  __shared__ int   mC[WAVES][16][TOPK];

  const int tid  = threadIdx.x;
  const int wave = tid >> 5;
  const int lane = tid & 31;
  const int nIdx = lane & 15;
  const int hi   = (lane >> 4) & 1;
  const int rowbase = blockIdx.x * 16;
  const int slice   = blockIdx.y;

  // Stage A operands: 16 rows of tgt and q, normalized, as bf16.
  for (int i = tid; i < 16 * 128; i += 128) {
    const int r = i >> 7, d4 = i & 127;
    const float st = inv[KQ + rowbase + r];
    const float sq = inv[KQ + BQ + rowbase + r];
    const float4 vt = ((const float4*)(tgt + (size_t)(rowbase + r) * DQ))[d4];
    const float4 vq = ((const float4*)(qf  + (size_t)(rowbase + r) * DQ))[d4];
    const int o = r * DQ + d4 * 4;
    uint2 pt, pq;
    pt.x = (unsigned)f2bf(vt.x * st) | ((unsigned)f2bf(vt.y * st) << 16);
    pt.y = (unsigned)f2bf(vt.z * st) | ((unsigned)f2bf(vt.w * st) << 16);
    pq.x = (unsigned)f2bf(vq.x * sq) | ((unsigned)f2bf(vq.y * sq) << 16);
    pq.y = (unsigned)f2bf(vq.z * sq) | ((unsigned)f2bf(vq.w * sq) << 16);
    *(uint2*)(sAt + o) = pt;
    *(uint2*)(sAq + o) = pq;
  }
  __syncthreads();

  Top5 top; top5_init(top);

  // A fragment base: lane holds row M = lane%16; K halves {d0+h, d0+16+h},
  // h = 0..7, with +8 K-offset for lanes 16-31 (ISA 7.12.2, 16-bit A 16x32).
  const unsigned short* arowT = sAt + nIdx * DQ + hi * 8;
  const unsigned short* arowQ = sAq + nIdx * DQ + hi * 8;

  for (int t = 0; t < TILES_PER_WAVE; ++t) {
    const int colbase = slice * SLICE_ROWS + (wave + WAVES * t) * 16;
    // B fragment: lane holds column N = lane%16 (bank row colbase+N),
    // K = d0 + 8*(lane>=16) + {0..7, 16..23} (matches A fragment layout).
    const unsigned short* brow =
        bankbf + (size_t)(colbase + nIdx) * DQ + hi * 8;

    v8f accT = {}; v8f accQ = {};
#pragma unroll
    for (int s = 0; s < 16; ++s) {
      const int d0 = s * 32;
      v8us b0 = *(const v8us*)(brow + d0);
      v8us b1 = *(const v8us*)(brow + d0 + 16);
      v16us bf = __builtin_shufflevector(b0, b1, 0,1,2,3,4,5,6,7,8,9,10,11,12,13,14,15);
      v8us a0 = *(const v8us*)(arowT + d0);
      v8us a1 = *(const v8us*)(arowT + d0 + 16);
      v16us af = __builtin_shufflevector(a0, a1, 0,1,2,3,4,5,6,7,8,9,10,11,12,13,14,15);
      v8us u0 = *(const v8us*)(arowQ + d0);
      v8us u1 = *(const v8us*)(arowQ + d0 + 16);
      v16us uf = __builtin_shufflevector(u0, u1, 0,1,2,3,4,5,6,7,8,9,10,11,12,13,14,15);
      accT = __builtin_amdgcn_wmma_f32_16x16x32_bf16(
          false, as_bf(af), false, as_bf(bf), (short)0, accT, false, false);
      accQ = __builtin_amdgcn_wmma_f32_16x16x32_bf16(
          false, as_bf(uf), false, as_bf(bf), (short)0, accQ, false, false);
    }

    // Transpose the 16x16 accumulators through LDS so one lane owns one row.
    // C layout: VGPR r, lane L -> M = r + 8*(L>=16), N = L%16.
    float* sc = sDist[wave];
#pragma unroll
    for (int r = 0; r < 8; ++r) {
      const int M = r + hi * 8;
      sc[M * 16 + nIdx]       = 2.0f - 2.0f * accT[r];
      sc[256 + M * 16 + nIdx] = 2.0f - 2.0f * accQ[r];
    }
    asm volatile("s_wait_dscnt 0" ::: "memory");  // per-wave LDS RAW fence

    if (lane < 16) {
#pragma unroll
      for (int c = 0; c < 16; ++c)
        top5_ins(top, sc[lane * 16 + c], sc[256 + lane * 16 + c], colbase + c);
    }
  }

  // Merge the 4 waves' candidates per row, emit per-(row, slice) top-5.
  if (lane < 16) {
    mT[wave][lane][0] = top.t0; mT[wave][lane][1] = top.t1; mT[wave][lane][2] = top.t2;
    mT[wave][lane][3] = top.t3; mT[wave][lane][4] = top.t4;
    mQ[wave][lane][0] = top.q0; mQ[wave][lane][1] = top.q1; mQ[wave][lane][2] = top.q2;
    mQ[wave][lane][3] = top.q3; mQ[wave][lane][4] = top.q4;
    mC[wave][lane][0] = top.c0; mC[wave][lane][1] = top.c1; mC[wave][lane][2] = top.c2;
    mC[wave][lane][3] = top.c3; mC[wave][lane][4] = top.c4;
  }
  __syncthreads();
  if (wave == 0 && lane < 16) {
    Top5 f; top5_init(f);
    for (int w = 0; w < WAVES; ++w)
      for (int j = 0; j < TOPK; ++j)
        top5_ins(f, mT[w][lane][j], mQ[w][lane][j], mC[w][lane][j]);
    const size_t base = ((size_t)(rowbase + lane) * NSLICE + slice) * TOPK;
    candT[base + 0] = f.t0; candT[base + 1] = f.t1; candT[base + 2] = f.t2;
    candT[base + 3] = f.t3; candT[base + 4] = f.t4;
    candQ[base + 0] = f.q0; candQ[base + 1] = f.q1; candQ[base + 2] = f.q2;
    candQ[base + 3] = f.q3; candQ[base + 4] = f.q4;
    candC[base + 0] = f.c0; candC[base + 1] = f.c1; candC[base + 2] = f.c2;
    candC[base + 3] = f.c3; candC[base + 4] = f.c4;
  }
}

// ---------------------------------------------------------------------------
// Kernel 4: zero the two outputs.
// ---------------------------------------------------------------------------
__global__ void zero_out_kernel(float* __restrict__ out) {
  if (threadIdx.x < 2) out[threadIdx.x] = 0.f;
}

// ---------------------------------------------------------------------------
// Kernel 5: per-row merge of 100 slice-local top-5s, loss/purity reduction.
// ---------------------------------------------------------------------------
__global__ __launch_bounds__(64)
void finalize_kernel(const float* __restrict__ candT,
                     const float* __restrict__ candQ,
                     const int*   __restrict__ candC,
                     const int*   __restrict__ labels,
                     const int*   __restrict__ qlabels,
                     float* __restrict__ out) {
  const int row = blockIdx.x * 64 + threadIdx.x;
  Top5 f; top5_init(f);
  const size_t base = (size_t)row * (NSLICE * TOPK);
  for (int e = 0; e < NSLICE * TOPK; ++e)
    top5_ins(f, candT[base + e], candQ[base + e], candC[base + e]);
  const int lab = labels[row];
  float ls = f.q0 + f.q1 + f.q2 + f.q3 + f.q4;
  float ps = (float)(qlabels[f.c0] == lab) + (float)(qlabels[f.c1] == lab) +
             (float)(qlabels[f.c2] == lab) + (float)(qlabels[f.c3] == lab) +
             (float)(qlabels[f.c4] == lab);
  const float scale = 1.0f / ((float)TOPK * (float)BQ);
  ls *= scale; ps *= scale;

  __shared__ float rl[64], rp[64];
  rl[threadIdx.x] = ls; rp[threadIdx.x] = ps;
  __syncthreads();
  for (int o = 32; o > 0; o >>= 1) {
    if (threadIdx.x < o) {
      rl[threadIdx.x] += rl[threadIdx.x + o];
      rp[threadIdx.x] += rp[threadIdx.x + o];
    }
    __syncthreads();
  }
  if (threadIdx.x == 0) {
    atomicAdd(&out[0], rl[0]);
    atomicAdd(&out[1], rp[0]);
  }
}

// ---------------------------------------------------------------------------
extern "C" void kernel_launch(void* const* d_in, const int* in_sizes, int n_in,
                              void* d_out, int out_size, void* d_ws, size_t ws_size,
                              hipStream_t stream) {
  const float* qf      = (const float*)d_in[0];  // q_feat        [B,D]
  const float* tgt     = (const float*)d_in[1];  // current_target[B,D]
  const float* bank    = (const float*)d_in[2];  // queue         [K,D]
  const int*   labels  = (const int*)d_in[3];    // [B]
  const int*   qlabels = (const int*)d_in[4];    // [K]
  // d_in[5] is k == 5 (hardcoded as TOPK)
  float* out = (float*)d_out;
  char*  ws  = (char*)d_ws;

  // Workspace layout.
  size_t off = 0;
  unsigned short* bankbf = (unsigned short*)(ws + off);
  off += (size_t)KQ * DQ * sizeof(unsigned short);            // 131 MB
  float* inv = (float*)(ws + off);
  off += (size_t)(KQ + 2 * BQ) * sizeof(float);
  off = (off + 255) & ~(size_t)255;
  const size_t candN = (size_t)BQ * NSLICE * TOPK;
  float* candT = (float*)(ws + off); off += candN * sizeof(float);
  float* candQ = (float*)(ws + off); off += candN * sizeof(float);
  int*   candC = (int*)(ws + off);   off += candN * sizeof(int);

  // 1) inverse norms: (K + 2B) rows, 8 waves/block.
  inv_norms_kernel<<<(KQ + 2 * BQ) / 8, 256, 0, stream>>>(bank, tgt, qf, inv);

  // 2) normalized bf16 bank (L2-resident working set).
  cvt_bank_kernel<<<(int)(((size_t)KQ * DQ / 4 + 255) / 256), 256, 0, stream>>>(
      bank, inv, bankbf);

  // 3) fused dual-GEMM + streaming top-5.
  dim3 grid(BQ / 16, NSLICE);
  meanshift_main_kernel<<<grid, WAVES * 32, 0, stream>>>(
      tgt, qf, bankbf, inv, candT, candQ, candC);

  // 4) + 5) final reduction.
  zero_out_kernel<<<1, 32, 0, stream>>>(out);
  finalize_kernel<<<BQ / 64, 64, 0, stream>>>(candT, candQ, candC, labels,
                                              qlabels, out);
}
